// GQAAttention_3410204033298
// MI455X (gfx1250) — matrix operand order repaired and checked by reading, not compiled, not run on hardware
//
#include <hip/hip_runtime.h>

// ---------------------------------------------------------------------------
// GQA attention block for MI455X (gfx1250, wave32, WMMA).
// All matmuls use v_wmma_f32_16x16x32_bf16 (fp32 accumulate).
// LDS staging uses gfx1250 async loads (GLOBAL_LOAD_ASYNC_TO_LDS_B128,
// tracked by ASYNCcnt) with double buffering in the GEMM.
// ---------------------------------------------------------------------------

typedef __bf16 bf16_t;
typedef __attribute__((ext_vector_type(16))) __bf16 v16bf;
typedef __attribute__((ext_vector_type(8)))  __bf16 v8bf;
typedef __attribute__((ext_vector_type(8)))  float  v8f;

union AB16 { v16bf v; v8bf h[2]; };

__device__ __forceinline__ bf16_t f2bf(float f) {
  unsigned u = __builtin_bit_cast(unsigned, f);
  unsigned r = (u + 0x7FFFu + ((u >> 16) & 1u)) >> 16;   // round-to-nearest-even
  return __builtin_bit_cast(__bf16, (unsigned short)r);
}

__device__ __forceinline__ v8f vzero8() {
  v8f z = {0.f, 0.f, 0.f, 0.f, 0.f, 0.f, 0.f, 0.f};
  return z;
}

// gfx1250 async copy: 16B global -> LDS, per-lane addresses, ASYNCcnt-tracked.
// LDS operand is the low 32 bits of the generic pointer (same lowering LLVM
// uses for generic->AS3 addrspacecast).
__device__ __forceinline__ void async_b128(const void* gptr, void* lptr) {
  unsigned lds = (unsigned)(unsigned long long)lptr;
  unsigned long long ga = (unsigned long long)gptr;
  asm volatile("global_load_async_to_lds_b128 %0, %1, off"
               :: "v"(lds), "v"(ga) : "memory");
}
__device__ __forceinline__ void wait_async0() {
  asm volatile("s_wait_asynccnt 0x0" ::: "memory");
}

// ------------------------------ f32 -> bf16 --------------------------------
__global__ __launch_bounds__(256) void cvt_f32_bf16(const float* __restrict__ in,
                                                    bf16_t* __restrict__ out, int n) {
  int i = blockIdx.x * 256 + threadIdx.x;
  if (i < n) out[i] = f2bf(in[i]);
}

// ------------------------------ WMMA GEMM ----------------------------------
// C[M,N] = A[M,K](bf16) * B[K,N](bf16) + bias[N], fp32 output.
// Block tile 128x128, 8 waves; wave tile 32x64 (2x4 WMMA tiles); K step 32.
// Async double-buffered LDS staging overlaps HBM traffic with WMMA issue.
__global__ __launch_bounds__(256) void gemm_bf16_kernel(
    const bf16_t* __restrict__ A, const bf16_t* __restrict__ B,
    const float* __restrict__ bias, float* __restrict__ C,
    int M, int N, int K)
{
  __shared__ alignas(16) bf16_t lA[2][128 * 32];
  __shared__ alignas(16) bf16_t lB[2][32 * 128];

  const int tid  = threadIdx.x;
  const int lane = tid & 31;
  const int w    = tid >> 5;
  const int wm   = w >> 1;          // 0..3 -> rows [wm*32, +32)
  const int wn   = w & 1;           // 0..1 -> cols [wn*64, +64)
  const int hf   = lane >> 4;       // lane half (A-layout K-octet select)
  const int lm   = lane & 15;

  const int n0 = blockIdx.x * 128;
  const int m0 = blockIdx.y * 128;

  const int ar = tid & 127, ac = (tid >> 7) * 16;   // A coop-load coords
  const int br = tid & 31,  bc = (tid >> 5) * 16;   // B coop-load coords

  v8f acc[2][4];
#pragma unroll
  for (int mt = 0; mt < 2; ++mt)
#pragma unroll
    for (int nt = 0; nt < 4; ++nt) acc[mt][nt] = vzero8();

  auto stage = [&](int buf, int k0) {
    const bf16_t* As = A + (size_t)(m0 + ar) * K + k0 + ac;
    const bf16_t* Bs = B + (size_t)(k0 + br) * N + n0 + bc;
    async_b128(As,     &lA[buf][ar * 32 + ac]);
    async_b128(As + 8, &lA[buf][ar * 32 + ac + 8]);
    async_b128(Bs,     &lB[buf][br * 128 + bc]);
    async_b128(Bs + 8, &lB[buf][br * 128 + bc + 8]);
    if (k0 + 64 < K) {                     // gfx1250 global_prefetch_b8
      __builtin_prefetch(As + 64, 0, 1);
      __builtin_prefetch(Bs + (size_t)64 * N, 0, 1);
    }
  };

  stage(0, 0);
  wait_async0();
  __syncthreads();

  int cur = 0;
  for (int k0 = 0; k0 < K; k0 += 32) {
    if (k0 + 32 < K) stage(cur ^ 1, k0 + 32);   // async fill of next buffer

    AB16 af[2];
#pragma unroll
    for (int mt = 0; mt < 2; ++mt) {
      int row = wm * 32 + mt * 16 + lm;
      af[mt].h[0] = *(const v8bf*)&lA[cur][row * 32 + hf * 8];
      af[mt].h[1] = *(const v8bf*)&lA[cur][row * 32 + hf * 8 + 16];
    }
#pragma unroll
    for (int nt = 0; nt < 4; ++nt) {
      v16bf bfr;
#pragma unroll
      for (int s = 0; s < 16; ++s)
        bfr[s] = lB[cur][((s & 8) * 2 + hf * 8 + (s & 7)) * 128 + wn * 64 + nt * 16 + lm];
#pragma unroll
      for (int mt = 0; mt < 2; ++mt)
        acc[mt][nt] = __builtin_amdgcn_wmma_f32_16x16x32_bf16(
            false, af[mt].v, false, bfr, (short)0, acc[mt][nt], false, false);
    }

    wait_async0();       // own async stores to LDS complete
    __syncthreads();     // everyone's complete + reads of cur finished
    cur ^= 1;
  }

#pragma unroll
  for (int mt = 0; mt < 2; ++mt)
#pragma unroll
    for (int nt = 0; nt < 4; ++nt) {
      int nc = n0 + wn * 64 + nt * 16 + lm;
      float bv = bias[nc];
#pragma unroll
      for (int r = 0; r < 8; ++r) {
        int mr = m0 + wm * 32 + mt * 16 + r + hf * 8;   // C/D layout: M = r + 8*half
        C[(size_t)mr * N + nc] = acc[mt][nt][r] + bv;
      }
    }
}

// --------------------------- RoPE + scatter --------------------------------
// qraw[4096][2048], kvraw[4096][1024] -> bf16 head-major qh/kh/vh.
// Q is pre-scaled by 1/sqrt(128).
__global__ __launch_bounds__(256) void rope_scatter_kernel(
    const float* __restrict__ qraw, const float* __restrict__ kvraw,
    bf16_t* __restrict__ qh, bf16_t* __restrict__ kh, bf16_t* __restrict__ vh)
{
  const int n = blockIdx.x * 256 + threadIdx.x;   // 0..3071
  const int m = blockIdx.y;                       // 0..4095
  const int t = m & 2047, b = m >> 11;
  const float LN1E4_64 = 9.210340371976184f / 64.0f;

  if (n < 2048) {
    int h = n >> 7, d = n & 127, f = d & 63;
    float val = qraw[(size_t)m * 2048 + n];
    float par = qraw[(size_t)m * 2048 + (n ^ 64)];
    float ang = (float)t * __expf(-(float)f * LN1E4_64);
    float s, c; __sincosf(ang, &s, &c);
    float o = (d < 64) ? (val * c - par * s) : (val * c + par * s);
    qh[(((size_t)(b * 16 + h)) * 2048 + t) * 128 + d] = f2bf(o * 0.08838834764831845f);
  } else {
    int c0 = n - 2048, kvh = c0 >> 8, j = c0 & 255;
    float val = kvraw[(size_t)m * 1024 + c0];
    if (j < 128) {
      int d = j, f = d & 63;
      float par = kvraw[(size_t)m * 1024 + (c0 ^ 64)];
      float ang = (float)t * __expf(-(float)f * LN1E4_64);
      float s, c; __sincosf(ang, &s, &c);
      float o = (d < 64) ? (val * c - par * s) : (val * c + par * s);
      kh[(((size_t)(b * 4 + kvh)) * 2048 + t) * 128 + d] = f2bf(o);
    } else {
      vh[(((size_t)(b * 4 + kvh)) * 2048 + t) * 128 + (j - 128)] = f2bf(val);
    }
  }
}

// -------------------------- Flash attention --------------------------------
// 4 waves/block; each wave owns 16 query rows of a 64-row Q tile.
// V is staged via async-to-LDS while the K transpose runs on VALU/DS.
__global__ __launch_bounds__(128) void flash_attn_kernel(
    const bf16_t* __restrict__ qh, const bf16_t* __restrict__ kh,
    const bf16_t* __restrict__ vh, bf16_t* __restrict__ attnb)
{
  __shared__ alignas(16) bf16_t lKT[128 * 64];     // [dim][token]
  __shared__ alignas(16) bf16_t lV [64 * 128];     // [token][dim]
  __shared__ alignas(16) bf16_t lP [4 * 16 * 64];  // per-wave P tile

  const int tid = threadIdx.x;
  const int w = tid >> 5, lane = tid & 31;
  const int hf = lane >> 4, lm = lane & 15;

  const int blk = blockIdx.x;
  const int qb  = blk & 31;
  const int h   = (blk >> 5) & 15;
  const int b   = blk >> 9;
  const int kvh = h >> 2;
  const int q0  = qb * 64;

  const bf16_t* qbase = qh + ((size_t)(b * 16 + h)  * 2048) * 128;
  const bf16_t* kbase = kh + ((size_t)(b * 4 + kvh) * 2048) * 128;
  const bf16_t* vbase = vh + ((size_t)(b * 4 + kvh) * 2048) * 128;

  // Q fragments (A layout), rows q0 + w*16 + lm, 4 chunks of K=32.
  AB16 qf[4];
  {
    const bf16_t* qp = qbase + (size_t)(q0 + w * 16 + lm) * 128;
#pragma unroll
    for (int dc = 0; dc < 4; ++dc) {
      int kb = dc * 32 + hf * 8;
      qf[dc].h[0] = *(const v8bf*)(qp + kb);
      qf[dc].h[1] = *(const v8bf*)(qp + kb + 16);
    }
  }

  v8f O[8];
#pragma unroll
  for (int o = 0; o < 8; ++o) O[o] = vzero8();
  float mrow[8], lrow[8];
#pragma unroll
  for (int r = 0; r < 8; ++r) { mrow[r] = -3.0e38f; lrow[r] = 0.f; }

  for (int k0 = 0; k0 <= q0; k0 += 64) {
    { // V: async copy to LDS; K: transpose through VALU/DS (overlaps async)
      int tt = tid >> 1;
      int dh = (tid & 1) * 64;
      const bf16_t* vs = vbase + (size_t)(k0 + tt) * 128 + dh;
      bf16_t* vd = &lV[tt * 128 + dh];
#pragma unroll
      for (int j = 0; j < 64; j += 8) async_b128(vs + j, vd + j);

      const bf16_t* ks = kbase + (size_t)(k0 + tt) * 128 + dh;
#pragma unroll
      for (int j = 0; j < 64; j += 8) {
        v8bf kv8 = *(const v8bf*)(ks + j);
#pragma unroll
        for (int e = 0; e < 8; ++e) lKT[(dh + j + e) * 64 + tt] = kv8[e];
      }
    }
    wait_async0();
    __syncthreads();

    // S = Q * K^T (16 rows x 64 cols per wave)
    v8f S[4];
#pragma unroll
    for (int nt = 0; nt < 4; ++nt) S[nt] = vzero8();
#pragma unroll
    for (int nt = 0; nt < 4; ++nt)
#pragma unroll
      for (int dc = 0; dc < 4; ++dc) {
        v16bf bk;
#pragma unroll
        for (int s = 0; s < 16; ++s)
          bk[s] = lKT[(dc * 32 + (s & 8) * 2 + hf * 8 + (s & 7)) * 64 + nt * 16 + lm];
        S[nt] = __builtin_amdgcn_wmma_f32_16x16x32_bf16(
            false, qf[dc].v, false, bk, (short)0, S[nt], false, false);
      }

    if (k0 == q0) {  // causal mask, diagonal block only
#pragma unroll
      for (int nt = 0; nt < 4; ++nt)
#pragma unroll
        for (int r = 0; r < 8; ++r) {
          int qg = w * 16 + r + hf * 8;
          int kg = nt * 16 + lm;
          if (kg > qg) S[nt][r] = -3.0e38f;
        }
    }

    // online softmax (row stats; row's 16 cols live in one 16-lane half)
    float alpha[8];
#pragma unroll
    for (int r = 0; r < 8; ++r) {
      float v = fmaxf(fmaxf(S[0][r], S[1][r]), fmaxf(S[2][r], S[3][r]));
#pragma unroll
      for (int off = 8; off >= 1; off >>= 1) v = fmaxf(v, __shfl_xor(v, off, 32));
      float mn = fmaxf(mrow[r], v);
      alpha[r] = __expf(mrow[r] - mn);
      mrow[r] = mn;
    }
    float rs[8];
#pragma unroll
    for (int r = 0; r < 8; ++r) rs[r] = 0.f;
#pragma unroll
    for (int nt = 0; nt < 4; ++nt)
#pragma unroll
      for (int r = 0; r < 8; ++r) {
        float pv = __expf(S[nt][r] - mrow[r]);
        S[nt][r] = pv;
        rs[r] += pv;
      }
#pragma unroll
    for (int r = 0; r < 8; ++r) {
#pragma unroll
      for (int off = 8; off >= 1; off >>= 1) rs[r] += __shfl_xor(rs[r], off, 32);
      lrow[r] = lrow[r] * alpha[r] + rs[r];
    }
#pragma unroll
    for (int o = 0; o < 8; ++o)
#pragma unroll
      for (int r = 0; r < 8; ++r) O[o][r] *= alpha[r];

    // P (C/D layout) -> LDS -> A layout (wave-private region, in-order DS)
    bf16_t* pp = &lP[w * 1024];
#pragma unroll
    for (int nt = 0; nt < 4; ++nt)
#pragma unroll
      for (int r = 0; r < 8; ++r)
        pp[(r + hf * 8) * 64 + nt * 16 + lm] = f2bf(S[nt][r]);

#pragma unroll
    for (int kc = 0; kc < 2; ++kc) {
      AB16 pa;
      const bf16_t* ppa = &lP[w * 1024 + lm * 64 + kc * 32 + hf * 8];
      pa.h[0] = *(const v8bf*)ppa;
      pa.h[1] = *(const v8bf*)(ppa + 16);
#pragma unroll
      for (int nt = 0; nt < 8; ++nt) {
        v16bf bv;
#pragma unroll
        for (int s = 0; s < 16; ++s)
          bv[s] = lV[(kc * 32 + (s & 8) * 2 + hf * 8 + (s & 7)) * 128 + nt * 16 + lm];
        O[nt] = __builtin_amdgcn_wmma_f32_16x16x32_bf16(
            false, pa.v, false, bv, (short)0, O[nt], false, false);
      }
    }
    __syncthreads();
  }

  // normalize and emit bf16 (row-major [B*T][C]) for the output projection
#pragma unroll
  for (int r = 0; r < 8; ++r) {
    float inv = 1.0f / lrow[r];
    int t = q0 + w * 16 + r + hf * 8;
    bf16_t* orow = attnb + (size_t)(b * 2048 + t) * 2048 + h * 128;
#pragma unroll
    for (int nt = 0; nt < 8; ++nt)
      orow[nt * 16 + lm] = f2bf(O[nt][r] * inv);
  }
}

// ------------------------------- launcher ----------------------------------
extern "C" void kernel_launch(void* const* d_in, const int* in_sizes, int n_in,
                              void* d_out, int out_size, void* d_ws, size_t ws_size,
                              hipStream_t stream) {
  const float* x   = (const float*)d_in[0];
  const float* Wq  = (const float*)d_in[1];
  const float* bq  = (const float*)d_in[2];
  const float* Wkv = (const float*)d_in[3];
  const float* bkv = (const float*)d_in[4];
  const float* Wo  = (const float*)d_in[5];
  const float* bo  = (const float*)d_in[6];
  float* out = (float*)d_out;

  char* p = (char*)d_ws;
  auto take = [&](size_t bytes) { char* q = p; p += (bytes + 255) & ~(size_t)255; return q; };

  bf16_t* xb    = (bf16_t*)take((size_t)4096 * 2048 * 2);
  bf16_t* wqb   = (bf16_t*)take((size_t)2048 * 2048 * 2);
  bf16_t* wkvb  = (bf16_t*)take((size_t)2048 * 1024 * 2);
  bf16_t* wob   = (bf16_t*)take((size_t)2048 * 2048 * 2);
  float*  qraw  = (float*) take((size_t)4096 * 2048 * 4);
  float*  kvraw = (float*) take((size_t)4096 * 1024 * 4);
  bf16_t* qh    = (bf16_t*)take((size_t)2 * 16 * 2048 * 128 * 2);
  bf16_t* kh    = (bf16_t*)take((size_t)2 * 4 * 2048 * 128 * 2);
  bf16_t* vh    = (bf16_t*)take((size_t)2 * 4 * 2048 * 128 * 2);
  bf16_t* attnb = (bf16_t*)take((size_t)4096 * 2048 * 2);

  cvt_f32_bf16<<<32768, 256, 0, stream>>>(x,   xb,   4096 * 2048);
  cvt_f32_bf16<<<16384, 256, 0, stream>>>(Wq,  wqb,  2048 * 2048);
  cvt_f32_bf16<<< 8192, 256, 0, stream>>>(Wkv, wkvb, 2048 * 1024);
  cvt_f32_bf16<<<16384, 256, 0, stream>>>(Wo,  wob,  2048 * 2048);

  gemm_bf16_kernel<<<dim3(16, 32), 256, 0, stream>>>(xb, wqb,  bq,  qraw,  4096, 2048, 2048);
  gemm_bf16_kernel<<<dim3( 8, 32), 256, 0, stream>>>(xb, wkvb, bkv, kvraw, 4096, 1024, 2048);

  rope_scatter_kernel<<<dim3(12, 4096), 256, 0, stream>>>(qraw, kvraw, qh, kh, vh);

  flash_attn_kernel<<<1024, 128, 0, stream>>>(qh, kh, vh, attnb);

  gemm_bf16_kernel<<<dim3(16, 32), 256, 0, stream>>>(attnb, wob, bo, out, 4096, 2048, 2048);
}
